// Block_9646496547176
// MI455X (gfx1250) — matrix operand order repaired and checked by reading, not compiled, not run on hardware
//
#include <hip/hip_runtime.h>
#include <math.h>

typedef __bf16 bf16;
typedef __attribute__((ext_vector_type(8)))  __bf16 v8bf;
typedef __attribute__((ext_vector_type(16))) __bf16 v16bf;
typedef __attribute__((ext_vector_type(8)))  float  v8f;

#define WMMA_BF16(a,b,c) __builtin_amdgcn_wmma_f32_16x16x32_bf16(false,(a),false,(b),(short)0,(c),false,false)

constexpr int Bb = 4, Tt = 2048, Cc = 1024, Hh = 16, Dd = 64, FFf = 2048;
constexpr int Mm = Bb * Tt; // 8192 rows

// ---- helpers ------------------------------------------------------------
// A-operand (16x32 bf16, MxK): lane m=lane&15, half=lane>>4; elem j=0..15 -> K=half*16+j.
// Two contiguous 16B loads.
__device__ inline v16bf load_afrag(const bf16* __restrict__ base, int ld, int m0, int k0, int nl, int hf) {
    const bf16* p = base + (size_t)(m0 + nl) * ld + k0 + hf * 16;
    v8bf lo = *(const v8bf*)p;
    v8bf hi = *(const v8bf*)(p + 8);
    return __builtin_shufflevector(lo, hi, 0,1,2,3,4,5,6,7,8,9,10,11,12,13,14,15);
}
// B-operand (32x16 bf16, KxN) from row-major BT[N][K]: lane n=lane&15, elem j -> K=half*16+j.
// One contiguous 32B load.
__device__ inline v16bf load_bfrag(const bf16* __restrict__ baseT, int ld, int n0, int k0, int nl, int hf) {
    return *(const v16bf*)(baseT + (size_t)(n0 + nl) * ld + k0 + hf * 16);
}

// ---- weight transpose + fp32->bf16 convert: out[N][K] = (bf16) in[K][N] --
__global__ void transpose_bf16_kernel(const float* __restrict__ in, bf16* __restrict__ out, int K, int N) {
    __shared__ float tile[32][33];
    int bx = blockIdx.x * 32, by = blockIdx.y * 32;
    int tx = threadIdx.x, ty = threadIdx.y; // 32 x 8
#pragma unroll
    for (int j = 0; j < 4; ++j)
        tile[ty + j * 8][tx] = in[(size_t)(by + ty + j * 8) * N + bx + tx];
    __syncthreads();
#pragma unroll
    for (int j = 0; j < 4; ++j)
        out[(size_t)(bx + ty + j * 8) * K + by + tx] = (bf16)tile[tx][ty + j * 8];
}

// ---- layernorm (one 256-thread block per row of C=1024) -> bf16 ---------
__global__ void ln_kernel(const float* __restrict__ x, const float* __restrict__ g,
                          const float* __restrict__ bta, bf16* __restrict__ out) {
    int row = blockIdx.x, tid = threadIdx.x;
    const float* xr = x + (size_t)row * Cc;
    float v[4], s = 0.f, s2 = 0.f;
#pragma unroll
    for (int j = 0; j < 4; ++j) { v[j] = xr[tid + j * 256]; s += v[j]; s2 += v[j] * v[j]; }
#pragma unroll
    for (int off = 16; off >= 1; off >>= 1) { s += __shfl_xor(s, off); s2 += __shfl_xor(s2, off); }
    __shared__ float rs[8], rs2[8];
    int w = tid >> 5, ln = tid & 31;
    if (ln == 0) { rs[w] = s; rs2[w] = s2; }
    __syncthreads();
    s = 0.f; s2 = 0.f;
#pragma unroll
    for (int i = 0; i < 8; ++i) { s += rs[i]; s2 += rs2[i]; }
    float mu = s * (1.0f / Cc);
    float var = s2 * (1.0f / Cc) - mu * mu;
    float r = rsqrtf(var + 1e-5f);
#pragma unroll
    for (int j = 0; j < 4; ++j) {
        int c = tid + j * 256;
        out[(size_t)row * Cc + c] = (bf16)((v[j] - mu) * r * g[c] + bta[c]);
    }
}

// ---- generic WMMA GEMM: D = A[M,K] * BT[N,K]^T, fused epilogues ----------
// Block: 128 threads = 4 waves (2x2), wave tile 64x64 (4x4 of 16x16), block tile 128x128.
// 16 WMMAs per 16 b128-loads per k-step -> 2x arithmetic intensity vs 64x32 tiles.
enum { EPI_QKV = 0, EPI_RESID = 1, EPI_GELU = 2 };

template <int EPI>
__global__ void gemm_kernel(const bf16* __restrict__ A, const bf16* __restrict__ BT,
                            int M, int N, int K,
                            const float* __restrict__ resid, const float* __restrict__ bias,
                            float* __restrict__ outf,
                            bf16* __restrict__ ob0, bf16* __restrict__ ob1, bf16* __restrict__ ob2) {
    int tid = threadIdx.x, w = tid >> 5, lane = tid & 31, nl = lane & 15, hf = lane >> 4;
    int m0 = blockIdx.y * 128 + (w & 1) * 64;
    int n0 = blockIdx.x * 128 + (w >> 1) * 64;

    v8f acc[4][4];
#pragma unroll
    for (int rt = 0; rt < 4; ++rt)
#pragma unroll
        for (int ct = 0; ct < 4; ++ct) acc[rt][ct] = (v8f)(0.f);

    for (int k0 = 0; k0 < K; k0 += 32) {
        if (k0 + 32 < K) { // light L2 prefetch of next strips
            __builtin_prefetch(A  + (size_t)(m0 + nl) * K + k0 + 32, 0, 1);
            __builtin_prefetch(BT + (size_t)(n0 + nl) * K + k0 + 32, 0, 1);
        }
        v16bf af[4], bfr[4];
#pragma unroll
        for (int rt = 0; rt < 4; ++rt) af[rt]  = load_afrag(A, K, m0 + rt * 16, k0, nl, hf);
#pragma unroll
        for (int ct = 0; ct < 4; ++ct) bfr[ct] = load_bfrag(BT, K, n0 + ct * 16, k0, nl, hf);
#pragma unroll
        for (int rt = 0; rt < 4; ++rt)
#pragma unroll
            for (int ct = 0; ct < 4; ++ct)
                acc[rt][ct] = WMMA_BF16(af[rt], bfr[ct], acc[rt][ct]);
    }

#pragma unroll
    for (int rt = 0; rt < 4; ++rt) {
#pragma unroll
        for (int ct = 0; ct < 4; ++ct) {
#pragma unroll
            for (int i = 0; i < 8; ++i) {
                int row = m0 + rt * 16 + i + 8 * hf;
                int col = n0 + ct * 16 + nl;
                float v = acc[rt][ct][i];
                if (EPI == EPI_QKV) {
                    int bb = row >> 11, t = row & (Tt - 1);
                    int sec = col >> 10, cc = col & (Cc - 1);
                    int h2 = cc >> 6, d2 = cc & (Dd - 1);
                    if (sec == 0)
                        ob0[(((size_t)bb * Hh + h2) * Tt + t) * Dd + d2] = (bf16)(v * 0.125f); // q * 1/sqrt(D)
                    else if (sec == 1)
                        ob1[(((size_t)bb * Hh + h2) * Tt + t) * Dd + d2] = (bf16)v;            // k
                    else
                        ob2[(((size_t)bb * Hh + h2) * Dd + d2) * Tt + t] = (bf16)v;            // v transposed
                } else if (EPI == EPI_RESID) {
                    size_t idx = (size_t)row * N + col;
                    outf[idx] = resid[idx] + v + bias[col];
                } else { // EPI_GELU (exact erf)
                    float gval = v + bias[col];
                    float r = 0.5f * gval * (1.0f + erff(gval * 0.70710678118654752f));
                    ob0[(size_t)row * N + col] = (bf16)r;
                }
            }
        }
    }
}

// ---- flash attention with permutation mask -------------------------------
// grid: (T/64, B*H), block 128 (4 waves); each wave owns 16 q-rows.
__global__ void attn_kernel(const bf16* __restrict__ qb, const bf16* __restrict__ kb,
                            const bf16* __restrict__ vtb, const int* __restrict__ perm,
                            bf16* __restrict__ y) {
    __shared__ __align__(16) bf16 pls[4 * 16 * 40]; // per-wave 16x32 P tile, stride 40
    int tid = threadIdx.x, w = tid >> 5, lane = tid & 31, nl = lane & 15, hf = lane >> 4;
    int bh = blockIdx.y, b = bh >> 4, h = bh & 15;
    int q0 = blockIdx.x * 64 + w * 16;
    bf16* pbase = pls + w * 640;

    // q A-fragments (K = D = 64 -> two 16x32 frags), q already scaled by 1/sqrt(D)
    const bf16* qrow = qb + ((size_t)bh * Tt + q0 + nl) * Dd;
    v16bf qf[2];
#pragma unroll
    for (int dk = 0; dk < 2; ++dk) {
        const bf16* p = qrow + dk * 32 + hf * 16;
        v8bf lo = *(const v8bf*)p;
        v8bf hi = *(const v8bf*)(p + 8);
        qf[dk] = __builtin_shufflevector(lo, hi, 0,1,2,3,4,5,6,7,8,9,10,11,12,13,14,15);
    }

    v8f o[4];
#pragma unroll
    for (int dt = 0; dt < 4; ++dt) o[dt] = (v8f)(0.f);
    float mst[8], lst[8];
#pragma unroll
    for (int i = 0; i < 8; ++i) { mst[i] = -3.0e38f; lst[i] = 0.f; }

    for (int k0 = 0; k0 < Tt; k0 += 32) {
        // scores: two 16x16 S tiles (k cols 0..15 and 16..31), K-dim = D = 64
        const bf16* kr0 = kb + ((size_t)bh * Tt + k0 + nl) * Dd;
        const bf16* kr1 = kr0 + (size_t)16 * Dd;
        v8f s0 = (v8f)(0.f), s1 = (v8f)(0.f);
#pragma unroll
        for (int dk = 0; dk < 2; ++dk) {
            v16bf kf0 = *(const v16bf*)(kr0 + dk * 32 + hf * 16);
            v16bf kf1 = *(const v16bf*)(kr1 + dk * 32 + hf * 16);
            s0 = WMMA_BF16(qf[dk], kf0, s0);
            s1 = WMMA_BF16(qf[dk], kf1, s1);
        }

        int pk0 = perm[k0 + nl];
        int pk1 = perm[k0 + 16 + nl];
        float p0[8], p1[8];
#pragma unroll
        for (int i = 0; i < 8; ++i) {
            int row = q0 + i + 8 * hf;
            float a = s0[i], c = s1[i];
            if (pk0 > row) a = -3.0e38f;
            if (pk1 > row) c = -3.0e38f;
            float tm = fmaxf(a, c);
            tm = fmaxf(tm, __shfl_xor(tm, 1));
            tm = fmaxf(tm, __shfl_xor(tm, 2));
            tm = fmaxf(tm, __shfl_xor(tm, 4));
            tm = fmaxf(tm, __shfl_xor(tm, 8));
            float mn = fmaxf(mst[i], tm);
            float sc = __expf(mst[i] - mn);
            float e0 = __expf(a - mn);
            float e1 = __expf(c - mn);
            float ls = e0 + e1;
            ls += __shfl_xor(ls, 1);
            ls += __shfl_xor(ls, 2);
            ls += __shfl_xor(ls, 4);
            ls += __shfl_xor(ls, 8);
            lst[i] = lst[i] * sc + ls;
            mst[i] = mn;
#pragma unroll
            for (int dt = 0; dt < 4; ++dt) o[dt][i] *= sc;
            p0[i] = e0; p1[i] = e1;
        }

        // C-layout -> A-layout transpose of P through LDS (wave-private region)
#pragma unroll
        for (int i = 0; i < 8; ++i) {
            pbase[(i + 8 * hf) * 40 + nl]      = (bf16)p0[i];
            pbase[(i + 8 * hf) * 40 + 16 + nl] = (bf16)p1[i];
        }
        __builtin_amdgcn_wave_barrier();
        asm volatile("" ::: "memory");
        v16bf pf;
        {
            const bf16* pr = pbase + nl * 40;
            v8bf lo = *(const v8bf*)(pr + hf * 16);
            v8bf hi = *(const v8bf*)(pr + 8 + hf * 16);
            pf = __builtin_shufflevector(lo, hi, 0,1,2,3,4,5,6,7,8,9,10,11,12,13,14,15);
        }
        __builtin_amdgcn_wave_barrier();

        // P @ V : B operand from v^T [B,H,D,T] (contiguous along k)
#pragma unroll
        for (int dt = 0; dt < 4; ++dt) {
            v16bf vf = *(const v16bf*)(vtb + ((size_t)bh * Dd + dt * 16 + nl) * Tt + k0 + hf * 16);
            o[dt] = WMMA_BF16(pf, vf, o[dt]);
        }
    }

    // normalize + store y[B,T,C] (bf16), C index = h*64 + d
#pragma unroll
    for (int i = 0; i < 8; ++i) {
        int row = q0 + i + 8 * hf;
        float inv = 1.0f / lst[i];
#pragma unroll
        for (int dt = 0; dt < 4; ++dt)
            y[((size_t)b * Tt + row) * Cc + h * Dd + dt * 16 + nl] = (bf16)(o[dt][i] * inv);
    }
}

// ---- host-side orchestration --------------------------------------------
extern "C" void kernel_launch(void* const* d_in, const int* in_sizes, int n_in,
                              void* d_out, int out_size, void* d_ws, size_t ws_size,
                              hipStream_t stream) {
    const float* x     = (const float*)d_in[0];
    const int*   perm  = (const int*)  d_in[1];
    const float* Wqkv  = (const float*)d_in[2];
    const float* Wproj = (const float*)d_in[3];
    const float* bproj = (const float*)d_in[4];
    const float* ln1g  = (const float*)d_in[5];
    const float* ln1b  = (const float*)d_in[6];
    const float* ln2g  = (const float*)d_in[7];
    const float* ln2b  = (const float*)d_in[8];
    const float* Wff1  = (const float*)d_in[9];
    const float* bff1  = (const float*)d_in[10];
    const float* Wff2  = (const float*)d_in[11];
    const float* bff2  = (const float*)d_in[12];
    float* out = (float*)d_out;

    char* ws = (char*)d_ws;
    size_t off = 0;
    auto take = [&](size_t bytes) { char* p = ws + off; off += bytes; return p; };
    bf16* WqkvT  = (bf16*)take((size_t)3 * Cc * Cc * 2); // [3C, C]
    bf16* WprojT = (bf16*)take((size_t)Cc * Cc * 2);     // [C, C]
    bf16* Wff1T  = (bf16*)take((size_t)FFf * Cc * 2);    // [FF, C]
    bf16* Wff2T  = (bf16*)take((size_t)Cc * FFf * 2);    // [C, FF]
    bf16* hbuf   = (bf16*)take((size_t)Mm * Cc * 2);     // ln1/ln2 output (reused)
    bf16* qbuf   = (bf16*)take((size_t)Mm * Cc * 2);     // [B,H,T,D]
    bf16* kbuf   = (bf16*)take((size_t)Mm * Cc * 2);     // [B,H,T,D]  (contiguous after qbuf)
    bf16* vtbuf  = (bf16*)take((size_t)Mm * Cc * 2);     // [B,H,D,T]
    bf16* ybuf   = (bf16*)take((size_t)Mm * Cc * 2);     // attn out [B,T,C]
    float* x1    = (float*)take((size_t)Mm * Cc * 4);    // residual after proj
    bf16* ffact  = qbuf;                                  // [M, FF] aliases q+k (32MB)

    dim3 tb(32, 8);
    // 1. weight transpose/convert
    transpose_bf16_kernel<<<dim3(3 * Cc / 32, Cc / 32), tb, 0, stream>>>(Wqkv, WqkvT, Cc, 3 * Cc);
    transpose_bf16_kernel<<<dim3(Cc / 32, Cc / 32),     tb, 0, stream>>>(Wproj, WprojT, Cc, Cc);
    transpose_bf16_kernel<<<dim3(FFf / 32, Cc / 32),    tb, 0, stream>>>(Wff1, Wff1T, Cc, FFf);
    transpose_bf16_kernel<<<dim3(Cc / 32, FFf / 32),    tb, 0, stream>>>(Wff2, Wff2T, FFf, Cc);

    // 2. LN1
    ln_kernel<<<Mm, 256, 0, stream>>>(x, ln1g, ln1b, hbuf);

    // 3. QKV projection, scatter into q / k / v^T
    gemm_kernel<EPI_QKV><<<dim3(3 * Cc / 128, Mm / 128), 128, 0, stream>>>(
        hbuf, WqkvT, Mm, 3 * Cc, Cc, nullptr, nullptr, nullptr, qbuf, kbuf, vtbuf);

    // 4. masked flash attention
    attn_kernel<<<dim3(Tt / 64, Bb * Hh), 128, 0, stream>>>(qbuf, kbuf, vtbuf, perm, ybuf);

    // 5. output projection + residual -> x1
    gemm_kernel<EPI_RESID><<<dim3(Cc / 128, Mm / 128), 128, 0, stream>>>(
        ybuf, WprojT, Mm, Cc, Cc, x, bproj, x1, nullptr, nullptr, nullptr);

    // 6. LN2
    ln_kernel<<<Mm, 256, 0, stream>>>(x1, ln2g, ln2b, hbuf);

    // 7. FF1 + GELU (exact erf)
    gemm_kernel<EPI_GELU><<<dim3(FFf / 128, Mm / 128), 128, 0, stream>>>(
        hbuf, Wff1T, Mm, FFf, Cc, nullptr, bff1, nullptr, ffact, nullptr, nullptr);

    // 8. FF2 + residual -> out
    gemm_kernel<EPI_RESID><<<dim3(Cc / 128, Mm / 128), 128, 0, stream>>>(
        ffact, Wff2T, Mm, Cc, FFf, x1, bff2, out, nullptr, nullptr, nullptr);
}
